// MultiScaleDeformableAttention2D_48086453846573
// MI455X (gfx1250) — compile-verified, matrix operand order checked by reading, not run on hardware
//
#include <hip/hip_runtime.h>
#include <math.h>

#define B_ 4
#define C_ 256
#define HEADS_ 8
#define S_Q 4096          // 64*64
#define HQ_ 64
#define WQ_ 64
#define NTOT (B_ * S_Q)   // 16384
#define PW 66             // padded width/height
#define PS (PW * PW)      // 4356 padded pixels per batch
#define KC 256            // contraction depth for every GEMM here

#if defined(__has_builtin)
#if __has_builtin(__builtin_amdgcn_sched_barrier)
#define SCHED_BARRIER() __builtin_amdgcn_sched_barrier(0)
#endif
#endif
#ifndef SCHED_BARRIER
#define SCHED_BARRIER()
#endif

typedef __bf16 bf16_t;
typedef __attribute__((ext_vector_type(16))) __bf16 v16bf;
typedef __attribute__((ext_vector_type(8)))  __bf16 v8bf;
typedef __attribute__((ext_vector_type(8)))  float  v8f;

__device__ __forceinline__ bf16_t f2bf(float f) {
  unsigned u = __builtin_bit_cast(unsigned, f);
  unsigned r = u + 0x7FFFu + ((u >> 16) & 1u);   // round-to-nearest-even
  unsigned short h = (unsigned short)(r >> 16);
  return __builtin_bit_cast(bf16_t, h);
}

__device__ __forceinline__ v16bf load_frag(const bf16_t* p) {
  // lane<16: K 0..7 at p, K 16..23 at p+16 ; lane>=16 caller pre-offsets by +8
  v8bf lo = *reinterpret_cast<const v8bf*>(p);
  v8bf hi = *reinterpret_cast<const v8bf*>(p + 16);
  v16bf r;
#pragma unroll
  for (int i = 0; i < 8; i++) { r[i] = lo[i]; r[i + 8] = hi[i]; }
  return r;
}

// ---------------- packing kernels ----------------

// LDS-tiled transpose: in (nb,256,S) f32 NCHW -> out (nb,S,256) bf16, coalesced both ways
__global__ void k_transpose_pack(const float* __restrict__ in, bf16_t* __restrict__ out,
                                 int S) {
  __shared__ float tile[32][33];
  const int b = blockIdx.z;
  const int s0 = blockIdx.x * 32, c0 = blockIdx.y * 32;
  const int tx = threadIdx.x, ty = threadIdx.y;   // 32 x 8
#pragma unroll
  for (int k = 0; k < 4; k++) {
    int c = c0 + ty + k * 8;
    tile[ty + k * 8][tx] = in[((long)b * 256 + c) * S + s0 + tx];
  }
  __syncthreads();
#pragma unroll
  for (int k = 0; k < 4; k++) {
    int s = s0 + ty + k * 8;
    out[((long)b * S + s) * 256 + c0 + tx] = f2bf(tile[tx][ty + k * 8]);
  }
}

__global__ void k_pack_bf16(const float* __restrict__ in, bf16_t* __restrict__ out, int n) {
  int i = blockIdx.x * blockDim.x + threadIdx.x;
  if (i < n) out[i] = f2bf(in[i]);
}

// in: (CO, 256, 3, 3) -> out: (9, CO, 256)  (tap-major, K contiguous)
__global__ void k_pack_conv_w(const float* __restrict__ in, bf16_t* __restrict__ out, int CO) {
  int i = blockIdx.x * blockDim.x + threadIdx.x;
  int total = CO * 256 * 9;
  if (i >= total) return;
  int c = i & 255;
  int o = (i >> 8) % CO;
  int tap = i / (256 * CO);
  int ky = tap / 3, kx = tap % 3;
  out[i] = f2bf(in[(((long)o * 256 + c) * 3 + ky) * 3 + kx]);
}

__global__ void k_zero(unsigned long long* __restrict__ p, long n) {
  long i = (long)blockIdx.x * blockDim.x + threadIdx.x;
  if (i < n) p[i] = 0ull;
}

// ---------------- WMMA GEMM: wave tile = 16(M) x 64(N), 4 accs, pinned pipeline ----------------
// D = A(256,256) * B(N,256)^T ; A,B bf16, K-contiguous rows.
// MODE 1: f32 out (N,256)
// MODE 2: f32 NCHW out with residual + BN + SiLU fused
// MODE 3: bf16 out into zero-padded (B,66,66,256) conv-input layout
template <int MODE>
__global__ __launch_bounds__(128) void k_gemm_wmma(
    const bf16_t* __restrict__ A, const bf16_t* __restrict__ Bm,
    void* __restrict__ Dst,
    const float* __restrict__ resid,
    const float* __restrict__ bn_g, const float* __restrict__ bn_b,
    const float* __restrict__ bn_m, const float* __restrict__ bn_v) {
  const int lane = threadIdx.x & 31;
  const int wave = threadIdx.x >> 5;
  const int lm = lane & 15;
  const int hi = lane >> 4;
  const int m0 = (blockIdx.x * 4 + wave) * 16;
  const int n0 = blockIdx.y * 64;

  const bf16_t* arow  = A + (long)(m0 + lm) * KC + hi * 8;
  const bf16_t* bbase = Bm + (long)(n0 + lm) * KC + hi * 8;

  v8f acc[4];
#pragma unroll
  for (int j = 0; j < 4; j++)
#pragma unroll
    for (int i = 0; i < 8; i++) acc[j][i] = 0.0f;

  // prologue: first chunk in flight
  v16bf af = load_frag(arow);
  v16bf bf[4];
#pragma unroll
  for (int j = 0; j < 4; j++) bf[j] = load_frag(bbase + (long)j * 16 * KC);

#pragma unroll
  for (int k0 = 0; k0 < KC; k0 += 32) {
    v16bf afc = af;
    v16bf bfc[4];
#pragma unroll
    for (int j = 0; j < 4; j++) bfc[j] = bf[j];
    if (k0 < KC - 32) {                      // issue next chunk's loads first
      af = load_frag(arow + k0 + 32);
#pragma unroll
      for (int j = 0; j < 4; j++) bf[j] = load_frag(bbase + (long)j * 16 * KC + k0 + 32);
    }
    SCHED_BARRIER();                          // keep loads above the WMMAs
#pragma unroll
    for (int j = 0; j < 4; j++)
      acc[j] = __builtin_amdgcn_wmma_f32_16x16x32_bf16(false, afc, false, bfc[j],
                                                       (short)0, acc[j], false, false);
    SCHED_BARRIER();
  }

  const int mbase = m0 + hi * 8;
#pragma unroll
  for (int j = 0; j < 4; j++) {
    const int n = n0 + j * 16 + lm;
    if (MODE == 1) {
      *reinterpret_cast<v8f*>((float*)Dst + (long)n * 256 + mbase) = acc[j];
    } else if (MODE == 2) {
      const int b = n >> 12;
      const int s = n & (S_Q - 1);
      float* dout = (float*)Dst;
#pragma unroll
      for (int r = 0; r < 8; r++) {
        int c = mbase + r;
        float x = acc[j][r] + resid[((long)b * C_ + c) * S_Q + s];
        float inv = rsqrtf(bn_v[c] + 1e-5f);
        float xn = (x - bn_m[c]) * (bn_g[c] * inv) + bn_b[c];
        dout[((long)b * C_ + c) * S_Q + s] = xn / (1.0f + expf(-xn));   // SiLU
      }
    } else {  // MODE 3: padded bf16
      const int b = n >> 12;
      const int s = n & (S_Q - 1);
      const int y = s >> 6, x = s & 63;
      const long pr = (long)b * PS + (long)(y + 1) * PW + (x + 1);
      v8bf o;
#pragma unroll
      for (int r = 0; r < 8; r++) o[r] = f2bf(acc[j][r]);
      *reinterpret_cast<v8bf*>((bf16_t*)Dst + pr * 256 + mbase) = o;
    }
  }
}

// ---------------- implicit-GEMM 3x3 conv on zero-padded input ----------------
// Wp: (9, CO, 256) bf16 ; Xpad: (B, 66, 66, 256) bf16 ; Out: (NTOT, CO) f32
// wave tile = 16(CO) x 64(x = one full image row), 4 accs, pinned pipeline per tap.
__global__ __launch_bounds__(128) void k_conv3x3_wmma(
    const bf16_t* __restrict__ Wp, const bf16_t* __restrict__ Xpad,
    const float* __restrict__ bias, float* __restrict__ Out, int CO) {
  const int lane = threadIdx.x & 31;
  const int wave = threadIdx.x >> 5;
  const int lm = lane & 15;
  const int hi = lane >> 4;
  const int m0 = (blockIdx.x * 4 + wave) * 16;
  const int row = blockIdx.y;       // 0 .. B*64-1
  const int b = row >> 6;
  const int y = row & 63;

  v8f acc[4];
#pragma unroll
  for (int j = 0; j < 4; j++)
#pragma unroll
    for (int i = 0; i < 8; i++) acc[j][i] = 0.0f;

#pragma unroll 1
  for (int tap = 0; tap < 9; tap++) {
    const int dy = tap / 3, dx = tap % 3;
    // lane (j,lm) reads padded pixel (y+dy, dx + j*16 + lm)
    const bf16_t* bbase = Xpad + ((long)(b * PW + y + dy) * PW + dx + lm) * 256 + hi * 8;
    const bf16_t* arow  = Wp + ((long)tap * CO + m0 + lm) * 256 + hi * 8;

    v16bf af = load_frag(arow);
    v16bf bf[4];
#pragma unroll
    for (int j = 0; j < 4; j++) bf[j] = load_frag(bbase + (long)j * 16 * 256);

#pragma unroll
    for (int k0 = 0; k0 < KC; k0 += 32) {
      v16bf afc = af;
      v16bf bfc[4];
#pragma unroll
      for (int j = 0; j < 4; j++) bfc[j] = bf[j];
      if (k0 < KC - 32) {
        af = load_frag(arow + k0 + 32);
#pragma unroll
        for (int j = 0; j < 4; j++) bf[j] = load_frag(bbase + (long)j * 16 * 256 + k0 + 32);
      }
      SCHED_BARRIER();                        // keep loads above the WMMAs
#pragma unroll
      for (int j = 0; j < 4; j++)
        acc[j] = __builtin_amdgcn_wmma_f32_16x16x32_bf16(false, afc, false, bfc[j],
                                                         (short)0, acc[j], false, false);
      SCHED_BARRIER();
    }
  }

  const int mbase = m0 + hi * 8;
#pragma unroll
  for (int j = 0; j < 4; j++) {
    const int n = b * S_Q + y * 64 + j * 16 + lm;
    v8f o;
#pragma unroll
    for (int r = 0; r < 8; r++) o[r] = acc[j][r] + bias[mbase + r];
    *reinterpret_cast<v8f*>(Out + (long)n * CO + mbase) = o;
  }
}

// ---------------- tanh offsets + softmax attention ----------------
// off_raw: (NTOT,256) ; logit_raw: (NTOT,128) -> off_out: (NTOT,8,32) ; attn_out: (NTOT,8,16)
__global__ void k_finalize(const float* __restrict__ off_raw,
                           const float* __restrict__ logit_raw,
                           float* __restrict__ off_out, float* __restrict__ attn_out) {
  int t = blockIdx.x * blockDim.x + threadIdx.x;   // (b*S + s)*8 + h
  if (t >= NTOT * HEADS_) return;
  int h = t & 7;
  long bs = t >> 3;
  const float* lp = logit_raw + bs * 128 + h * 16;
  float buf[16];
  float mx = -1e30f;
#pragma unroll
  for (int i = 0; i < 16; i++) { buf[i] = lp[i]; mx = fmaxf(mx, buf[i]); }
  float sum = 0.f;
#pragma unroll
  for (int i = 0; i < 16; i++) { buf[i] = expf(buf[i] - mx); sum += buf[i]; }
  float inv = 1.0f / sum;
  float* ao = attn_out + (long)t * 16;
#pragma unroll
  for (int i = 0; i < 16; i++) ao[i] = buf[i] * inv;
  const float* op = off_raw + bs * 256 + h * 32;
  float* oo = off_out + (long)t * 32;
#pragma unroll
  for (int j = 0; j < 32; j++) oo[j] = tanhf(op[j]) * 0.25f;
}

// ---------------- deformable bilinear sampling + weighted aggregation ----------------
// one wave per (b,h,s); lane = head-dim channel (HD=32)
__global__ void k_sample(const float* __restrict__ v_sc,    // rows (21760, 256) f32
                         const float* __restrict__ off,     // (NTOT,8,32)
                         const float* __restrict__ attn,    // (NTOT,8,16)
                         bf16_t* __restrict__ agg) {        // (NTOT,256) bf16
  const int lane = threadIdx.x & 31;
  const int w = blockIdx.x * (blockDim.x >> 5) + (threadIdx.x >> 5);
  const int s = w & (S_Q - 1);
  const int h = (w >> 12) & 7;
  const int b = w >> 15;
  const int yi = s >> 6, xi = s & 63;
  const float refx = -1.0f + 2.0f * xi * (1.0f / 63.0f);   // align_corners linspace
  const float refy = -1.0f + 2.0f * yi * (1.0f / 63.0f);
  const long bs = (long)b * S_Q + s;
  const float* offp = off + (bs * 8 + h) * 32;
  const float* attp = attn + (bs * 8 + h) * 16;
  const int Wl_arr[4] = {64, 32, 16, 8};
  const int rowoff[4] = {0, 16384, 20480, 21504};
  float acc = 0.0f;
#pragma unroll
  for (int l = 0; l < 4; l++) {
    const int Wl = Wl_arr[l];
    const int Sl = Wl * Wl;
    const float* vrow = v_sc + ((long)(rowoff[l] + b * Sl)) * 256 + h * 32 + lane;
    const float sc = 0.5f * (float)(Wl - 1);
#pragma unroll
    for (int p = 0; p < 4; p++) {
      float gx = refx + offp[(l * 4 + p) * 2 + 0];
      float gy = refy + offp[(l * 4 + p) * 2 + 1];
      float x = (gx + 1.0f) * sc;
      float yq = (gy + 1.0f) * sc;
      float xf = floorf(x), yf = floorf(yq);
      int ix = (int)xf, iy = (int)yf;
      float wx1 = x - xf, wx0 = 1.0f - wx1;
      float wy1 = yq - yf, wy0 = 1.0f - wy1;
      float aw = attp[l * 4 + p];
      float sv = 0.0f;
      if (ix >= 0 && ix < Wl && iy >= 0 && iy < Wl)                 sv += wx0 * wy0 * vrow[(iy * Wl + ix) * 256];
      if (ix + 1 >= 0 && ix + 1 < Wl && iy >= 0 && iy < Wl)         sv += wx1 * wy0 * vrow[(iy * Wl + ix + 1) * 256];
      if (ix >= 0 && ix < Wl && iy + 1 >= 0 && iy + 1 < Wl)         sv += wx0 * wy1 * vrow[((iy + 1) * Wl + ix) * 256];
      if (ix + 1 >= 0 && ix + 1 < Wl && iy + 1 >= 0 && iy + 1 < Wl) sv += wx1 * wy1 * vrow[((iy + 1) * Wl + ix + 1) * 256];
      acc += aw * sv;
    }
  }
  agg[bs * 256 + h * 32 + lane] = f2bf(acc);
}

// ---------------- host launch ----------------
extern "C" void kernel_launch(void* const* d_in, const int* in_sizes, int n_in,
                              void* d_out, int out_size, void* d_ws, size_t ws_size,
                              hipStream_t stream) {
  (void)in_sizes; (void)n_in; (void)out_size; (void)ws_size;
  const float* query  = (const float*)d_in[0];
  const float* feats[4] = {(const float*)d_in[1], (const float*)d_in[2],
                           (const float*)d_in[3], (const float*)d_in[4]};
  const float* q_w    = (const float*)d_in[5];
  const float* v_w    = (const float*)d_in[6];
  const float* out_w  = (const float*)d_in[7];
  const float* off_w  = (const float*)d_in[8];
  const float* off_b  = (const float*)d_in[9];
  const float* attn_w = (const float*)d_in[10];
  const float* attn_b = (const float*)d_in[11];
  const float* bn_g   = (const float*)d_in[12];
  const float* bn_b   = (const float*)d_in[13];
  const float* bn_m   = (const float*)d_in[14];
  const float* bn_v   = (const float*)d_in[15];

  char* ws = (char*)d_ws;
  size_t off = 0;
  auto alloc = [&](size_t bytes) -> char* {
    char* p = ws + off;
    off += (bytes + 255) & ~(size_t)255;
    return p;
  };
  bf16_t* qin    = (bf16_t*)alloc((size_t)NTOT * 256 * 2);
  bf16_t* fin    = (bf16_t*)alloc((size_t)21760 * 256 * 2);
  bf16_t* qwb    = (bf16_t*)alloc((size_t)65536 * 2);
  bf16_t* vwb    = (bf16_t*)alloc((size_t)262144 * 2);
  bf16_t* owb    = (bf16_t*)alloc((size_t)65536 * 2);
  bf16_t* offwb  = (bf16_t*)alloc((size_t)9 * 256 * 256 * 2);
  bf16_t* attwb  = (bf16_t*)alloc((size_t)9 * 128 * 256 * 2);
  bf16_t* qpad   = (bf16_t*)alloc((size_t)B_ * PS * 256 * 2);   // zero-padded conv input
  float*  offraw = (float*)alloc((size_t)NTOT * 256 * 4);
  float*  lraw   = (float*)alloc((size_t)NTOT * 128 * 4);
  float*  offf   = (float*)alloc((size_t)NTOT * 8 * 32 * 4);
  float*  attf   = (float*)alloc((size_t)NTOT * 8 * 16 * 4);
  float*  vsc    = (float*)alloc((size_t)21760 * 256 * 4);
  bf16_t* agg    = (bf16_t*)alloc((size_t)NTOT * 256 * 2);

  const int Sl[4]     = {4096, 1024, 256, 64};
  const int rowoff[4] = {0, 16384, 20480, 21504};

  // pack activations (NCHW f32 -> (b,s,c) bf16) via LDS-tiled transpose
  k_transpose_pack<<<dim3(S_Q / 32, 8, B_), dim3(32, 8), 0, stream>>>(query, qin, S_Q);
  for (int l = 0; l < 4; l++) {
    k_transpose_pack<<<dim3(Sl[l] / 32, 8, B_), dim3(32, 8), 0, stream>>>(
        feats[l], fin + (size_t)rowoff[l] * 256, Sl[l]);
  }
  // pack weights
  k_pack_bf16<<<(65536 + 255) / 256, 256, 0, stream>>>(q_w, qwb, 65536);
  k_pack_bf16<<<(262144 + 255) / 256, 256, 0, stream>>>(v_w, vwb, 262144);
  k_pack_bf16<<<(65536 + 255) / 256, 256, 0, stream>>>(out_w, owb, 65536);
  k_pack_conv_w<<<((9 * 256 * 256) + 255) / 256, 256, 0, stream>>>(off_w, offwb, 256);
  k_pack_conv_w<<<((9 * 128 * 256) + 255) / 256, 256, 0, stream>>>(attn_w, attwb, 128);

  // zero the padded conv-input buffer (border must be 0 every call)
  {
    long n8 = ((long)B_ * PS * 256 * 2) / 8;
    k_zero<<<(unsigned)((n8 + 255) / 256), 256, 0, stream>>>((unsigned long long*)qpad, n8);
  }

  dim3 blk(128);
  // q projection -> bf16 written into padded layout (MODE 3)
  k_gemm_wmma<3><<<dim3(256 / 64, NTOT / 64), blk, 0, stream>>>(
      qwb, qin, qpad, nullptr, nullptr, nullptr, nullptr, nullptr);
  // 3x3 convs as implicit GEMM on padded input (branch-free)
  k_conv3x3_wmma<<<dim3(256 / 64, B_ * 64), blk, 0, stream>>>(offwb, qpad, off_b, offraw, 256);
  k_conv3x3_wmma<<<dim3(128 / 64, B_ * 64), blk, 0, stream>>>(attwb, qpad, attn_b, lraw, 128);
  // tanh*scale + softmax
  k_finalize<<<(NTOT * 8 + 255) / 256, 256, 0, stream>>>(offraw, lraw, offf, attf);
  // value projections per level -> f32 (b,s,c)
  for (int l = 0; l < 4; l++) {
    int Nl = B_ * Sl[l];
    k_gemm_wmma<1><<<dim3(256 / 64, Nl / 64), blk, 0, stream>>>(
        vwb + (size_t)l * 65536, fin + (size_t)rowoff[l] * 256,
        vsc + (size_t)rowoff[l] * 256,
        nullptr, nullptr, nullptr, nullptr, nullptr);
  }
  // deformable sampling + aggregation -> bf16 (b,s,c)
  k_sample<<<(NTOT * 8) / 8, 256, 0, stream>>>(vsc, offf, attf, agg);
  // output projection + residual + BN + SiLU -> NCHW f32 d_out
  k_gemm_wmma<2><<<dim3(256 / 64, NTOT / 64), blk, 0, stream>>>(
      owb, agg, d_out, query, bn_g, bn_b, bn_m, bn_v);
}
